// Head_70437463654801
// MI455X (gfx1250) — matrix operand order repaired and checked by reading, not compiled, not run on hardware
//
#include <hip/hip_runtime.h>
#include <hip/hip_bf16.h>

// MI455X / gfx1250 fused "Head" attention pipeline, bf16 WMMA everywhere.
// T = A = 512, H = 64, B*C = 128 slices.

#define T_DIM   512
#define A_DIM   512
#define H_DIM   64
#define NSLICE  128
#define SCALE   0.125f   // 64^-0.5

typedef __bf16 bf16_t;
typedef bf16_t bf16x8  __attribute__((ext_vector_type(8)));
typedef bf16_t bf16x16 __attribute__((ext_vector_type(16)));
typedef float  f32x4   __attribute__((ext_vector_type(4)));
typedef float  f32x8   __attribute__((ext_vector_type(8)));

__device__ __forceinline__ f32x8 zero8() {
  f32x8 z = {0.f, 0.f, 0.f, 0.f, 0.f, 0.f, 0.f, 0.f};
  return z;
}

// D = A*B + C, 16x16x32 bf16 -> f32
__device__ __forceinline__ f32x8 wmma_bf16(bf16x16 a, bf16x16 b, f32x8 c) {
  return __builtin_amdgcn_wmma_f32_16x16x32_bf16(
      /*neg_a=*/false, a, /*neg_b=*/false, b,
      /*c_mod=*/(short)0, c, /*reuse_a=*/false, /*reuse_b=*/false);
}

// A-fragment (16x32 bf16) from a row-major bf16 matrix, leading dim `ld`.
// Lane l: row = l&15; K chunks [hi..hi+7] and [hi+16..hi+23], hi = (l>>4)*8.
__device__ __forceinline__ bf16x16 load_a_bf16(const bf16_t* __restrict__ base,
                                               int ld, int lane) {
  const int r  = lane & 15;
  const int hi = (lane >> 4) << 3;
  const bf16_t* p = base + r * ld + hi;
  bf16x8 c0 = *(const bf16x8*)(p);
  bf16x8 c1 = *(const bf16x8*)(p + 16);
  return __builtin_shufflevector(c0, c1, 0, 1, 2, 3, 4, 5, 6, 7,
                                 8, 9, 10, 11, 12, 13, 14, 15);
}

// Same A-fragment but sourced from f32 memory with on-the-fly bf16 convert.
__device__ __forceinline__ bf16x16 load_a_f32(const float* __restrict__ base,
                                              int ld, int lane) {
  const int r  = lane & 15;
  const int hi = (lane >> 4) << 3;
  const float* p = base + r * ld + hi;
  f32x4 c0 = *(const f32x4*)(p);
  f32x4 c1 = *(const f32x4*)(p + 4);
  f32x4 c2 = *(const f32x4*)(p + 16);
  f32x4 c3 = *(const f32x4*)(p + 20);
  bf16x16 v;
#pragma unroll
  for (int j = 0; j < 4; ++j) {
    v[j]      = (bf16_t)c0[j];
    v[4 + j]  = (bf16_t)c1[j];
    v[8 + j]  = (bf16_t)c2[j];
    v[12 + j] = (bf16_t)c3[j];
  }
  return v;
}

// B-fragment (32x16 bf16) from an n-major matrix mat[n][k], leading dim `ld`.
// Lane l: col n = l&15; K chunks [kb..kb+7] and [kb+16..kb+23], kb = (l>>4)*8.
__device__ __forceinline__ bf16x16 load_b_bf16(const bf16_t* __restrict__ base,
                                               int ld, int lane) {
  const int n  = lane & 15;
  const int kb = (lane >> 4) << 3;
  const bf16_t* p = base + n * ld + kb;
  bf16x8 c0 = *(const bf16x8*)(p);
  bf16x8 c1 = *(const bf16x8*)(p + 16);
  return __builtin_shufflevector(c0, c1, 0, 1, 2, 3, 4, 5, 6, 7,
                                 8, 9, 10, 11, 12, 13, 14, 15);
}

// ---------------------------------------------------------------------------
// K0: transpose the five [A,H] f32 weights into bf16 [H][A] (B-fragment form).
// wt layout: [Wp, Wk, Wki, Wv, Wpi], each 64x512.
// ---------------------------------------------------------------------------
__global__ void wtrans_kernel(const float* __restrict__ Wp,
                              const float* __restrict__ Wk,
                              const float* __restrict__ Wki,
                              const float* __restrict__ Wv,
                              const float* __restrict__ Wpi,
                              bf16_t* __restrict__ wt) {
  const int m = blockIdx.x;
  const float* src = (m == 0) ? Wp : (m == 1) ? Wk : (m == 2) ? Wki
                    : (m == 3) ? Wv : Wpi;
  bf16_t* dst = wt + (size_t)m * H_DIM * A_DIM;
  for (int i = threadIdx.x; i < A_DIM * H_DIM; i += blockDim.x) {
    const int a = i >> 6;   // src row   (A)
    const int h = i & 63;   // src col   (H)
    dst[h * A_DIM + a] = (bf16_t)src[i];
  }
}

// ---------------------------------------------------------------------------
// K1: projections. One projection per wave (wave id == projection id), each
// wave owns a 32-row strip: acc = 4 h-tiles x 2 m-tiles = 64 VGPRs, so
// occupancy (not a single dependence chain) hides the streaming latency and
// every weight B-fragment load feeds two WMMAs.
//   p=0: pos = idx_i@Wp+bp      p=1: key   = idx_t@Wk+bk
//   p=2: key_i = idx_ir@Wki+bki p=3: val   = idx_i@Wv+bv (stored [H][T])
// ---------------------------------------------------------------------------
__global__ __launch_bounds__(128) void proj_kernel(
    const float* __restrict__ idx_i, const float* __restrict__ idx_t,
    const float* __restrict__ idx_ir,
    const bf16_t* __restrict__ WT,   // [4][64][512] : WpT, WkT, WkiT, WvT
    const float* __restrict__ bp, const float* __restrict__ bk,
    const float* __restrict__ bki, const float* __restrict__ bv,
    bf16_t* __restrict__ posb, bf16_t* __restrict__ keyb,
    bf16_t* __restrict__ keyib, bf16_t* __restrict__ valT) {
  const int lane  = threadIdx.x & 31;
  const int p     = threadIdx.x >> 5;          // projection for this wave
  const int slice = blockIdx.x;
  const int row0  = blockIdx.y * 32;

  const float* in   = (p == 1) ? idx_t : (p == 2) ? idx_ir : idx_i;
  const float* bias = (p == 0) ? bp : (p == 1) ? bk : (p == 2) ? bki : bv;
  const bf16_t* wt  = WT + (size_t)p * H_DIM * A_DIM;

  const float* Abase = in + (size_t)slice * T_DIM * A_DIM + (size_t)row0 * A_DIM;

  f32x8 acc[4][2];
#pragma unroll
  for (int h = 0; h < 4; ++h) { acc[h][0] = zero8(); acc[h][1] = zero8(); }

#pragma unroll 2
  for (int ks = 0; ks < 16; ++ks) {
    const int k0 = ks * 32;
    if (ks + 1 < 16) {   // pull next K-chunk toward L2 -> global_prefetch_b8
      __builtin_prefetch(Abase + k0 + 32, 0, 1);
      __builtin_prefetch(Abase + 16 * A_DIM + k0 + 32, 0, 1);
    }
    bf16x16 a0 = load_a_f32(Abase + k0, A_DIM, lane);                // rows 0..15
    bf16x16 a1 = load_a_f32(Abase + 16 * A_DIM + k0, A_DIM, lane);   // rows 16..31
#pragma unroll
    for (int h = 0; h < 4; ++h) {
      bf16x16 b = load_b_bf16(wt + h * 16 * A_DIM + k0, A_DIM, lane);
      acc[h][0] = wmma_bf16(a0, b, acc[h][0]);
      acc[h][1] = wmma_bf16(a1, b, acc[h][1]);
    }
  }

  const int n  = lane & 15;
  const int mb = (lane >> 4) << 3;
  if (p == 3) {
    // val, stored transposed: valT[h][t]
    bf16_t* vT = valT + (size_t)slice * H_DIM * T_DIM;
#pragma unroll
    for (int h = 0; h < 4; ++h) {
      const int col = h * 16 + n;
      const float bb = bias[col];
#pragma unroll
      for (int m2 = 0; m2 < 2; ++m2)
#pragma unroll
        for (int r = 0; r < 8; ++r) {
          const int row = row0 + m2 * 16 + mb + r;
          vT[(size_t)col * T_DIM + row] = (bf16_t)(acc[h][m2][r] + bb);
        }
    }
  } else {
    bf16_t* dst = ((p == 0) ? posb : (p == 1) ? keyb : keyib)
                  + (size_t)slice * T_DIM * H_DIM + (size_t)row0 * H_DIM;
#pragma unroll
    for (int h = 0; h < 4; ++h) {
      const int col = h * 16 + n;
      const float bb = bias[col];
#pragma unroll
      for (int m2 = 0; m2 < 2; ++m2)
#pragma unroll
        for (int r = 0; r < 8; ++r)
          dst[(m2 * 16 + mb + r) * H_DIM + col] = (bf16_t)(acc[h][m2][r] + bb);
    }
  }
}

// ---------------------------------------------------------------------------
// K2: wei1 = (pos . key^T) * scale, then wei2 = wei1 @ Wpi + bpi.
// Each wave builds its 16x512 wei1 strip tile-by-tile into a private LDS
// strip (D-layout -> A-layout conversion), then contracts it against WpiT.
// Eliminates the 64 MiB [T,T] intermediate entirely.
// ---------------------------------------------------------------------------
__global__ __launch_bounds__(128) void pk_pi_kernel(
    const bf16_t* __restrict__ posb, const bf16_t* __restrict__ keyb,
    const bf16_t* __restrict__ WTpi, const float* __restrict__ bpi,
    bf16_t* __restrict__ wei2b) {
  __shared__ __align__(16) bf16_t lds[4][16][T_DIM];   // 64 KB
  const int lane  = threadIdx.x & 31;
  const int wave  = threadIdx.x >> 5;
  const int slice = blockIdx.x;
  const int row0  = blockIdx.y * 64 + wave * 16;
  const int n  = lane & 15;
  const int mb = (lane >> 4) << 3;

  const bf16_t* pos = posb + (size_t)slice * T_DIM * H_DIM + (size_t)row0 * H_DIM;
  const bf16_t* key = keyb + (size_t)slice * T_DIM * H_DIM;
  bf16_t* strip = &lds[wave][0][0];

  bf16x16 a0 = load_a_bf16(pos, H_DIM, lane);        // K = 0..31 of H
  bf16x16 a1 = load_a_bf16(pos + 32, H_DIM, lane);   // K = 32..63 of H

  for (int nt = 0; nt < 32; ++nt) {                  // 512 cols of wei1
    f32x8 acc = zero8();
    acc = wmma_bf16(a0, load_b_bf16(key + nt * 16 * H_DIM,      H_DIM, lane), acc);
    acc = wmma_bf16(a1, load_b_bf16(key + nt * 16 * H_DIM + 32, H_DIM, lane), acc);
#pragma unroll
    for (int r = 0; r < 8; ++r)
      strip[(mb + r) * T_DIM + nt * 16 + n] = (bf16_t)(acc[r] * SCALE);
  }
  __syncthreads();

  bf16_t* w2 = wei2b + (size_t)slice * T_DIM * H_DIM + (size_t)row0 * H_DIM;
#pragma unroll
  for (int ht = 0; ht < 4; ++ht) {
    f32x8 acc = zero8();
#pragma unroll 4
    for (int ks = 0; ks < 16; ++ks) {
      bf16x16 a = load_a_bf16(strip + ks * 32, T_DIM, lane);
      bf16x16 b = load_b_bf16(WTpi + ht * 16 * A_DIM + ks * 32, A_DIM, lane);
      acc = wmma_bf16(a, b, acc);
    }
    const float bb = bpi[ht * 16 + n];
#pragma unroll
    for (int r = 0; r < 8; ++r)
      w2[(mb + r) * H_DIM + ht * 16 + n] = (bf16_t)(acc[r] + bb);
  }
}

// ---------------------------------------------------------------------------
// K3: wei3 = (wei2 . key_i^T) * scale, causal mask, softmax, out = P @ val.
// A wave holds its full 16x512 row strip in 32 f32 accumulator tiles so the
// row max/sum is exact; cross-lane reduction is wave32-correct (rows live in
// 16-lane halves -> shfl_xor 8/4/2/1). P round-trips through LDS as bf16 to
// become an A-fragment for the final WMMA against val^T.
// ---------------------------------------------------------------------------
__global__ __launch_bounds__(128) void attn_kernel(
    const bf16_t* __restrict__ wei2b, const bf16_t* __restrict__ keyib,
    const bf16_t* __restrict__ valT, float* __restrict__ out) {
  __shared__ __align__(16) bf16_t lds[4][16][T_DIM];   // 64 KB
  const int lane  = threadIdx.x & 31;
  const int wave  = threadIdx.x >> 5;
  const int slice = blockIdx.x;
  const int row0  = blockIdx.y * 64 + wave * 16;
  const int n  = lane & 15;
  const int mb = (lane >> 4) << 3;

  const bf16_t* w2 = wei2b + (size_t)slice * T_DIM * H_DIM + (size_t)row0 * H_DIM;
  const bf16_t* ki = keyib + (size_t)slice * T_DIM * H_DIM;

  bf16x16 a0 = load_a_bf16(w2, H_DIM, lane);
  bf16x16 a1 = load_a_bf16(w2 + 32, H_DIM, lane);

  f32x8 acc[32];
#pragma unroll
  for (int nt = 0; nt < 32; ++nt) acc[nt] = zero8();
#pragma unroll 8
  for (int nt = 0; nt < 32; ++nt) {
    acc[nt] = wmma_bf16(a0, load_b_bf16(ki + nt * 16 * H_DIM,      H_DIM, lane), acc[nt]);
    acc[nt] = wmma_bf16(a1, load_b_bf16(ki + nt * 16 * H_DIM + 32, H_DIM, lane), acc[nt]);
  }

  // scale + causal mask + row max
  float mx[8];
#pragma unroll
  for (int r = 0; r < 8; ++r) mx[r] = -3.0e38f;
#pragma unroll
  for (int nt = 0; nt < 32; ++nt) {
    const int col = nt * 16 + n;
#pragma unroll
    for (int r = 0; r < 8; ++r) {
      const int row = row0 + mb + r;
      float v = acc[nt][r] * SCALE;
      v = (col <= row) ? v : -3.0e38f;
      acc[nt][r] = v;
      mx[r] = fmaxf(mx[r], v);
    }
  }
#pragma unroll
  for (int r = 0; r < 8; ++r) {
#pragma unroll
    for (int off = 8; off > 0; off >>= 1)
      mx[r] = fmaxf(mx[r], __shfl_xor(mx[r], off, 32));
  }

  // exp + row sum
  float sm[8];
#pragma unroll
  for (int r = 0; r < 8; ++r) sm[r] = 0.f;
#pragma unroll
  for (int nt = 0; nt < 32; ++nt) {
#pragma unroll
    for (int r = 0; r < 8; ++r) {
      const float e = __expf(acc[nt][r] - mx[r]);
      acc[nt][r] = e;
      sm[r] += e;
    }
  }
#pragma unroll
  for (int r = 0; r < 8; ++r) {
#pragma unroll
    for (int off = 8; off > 0; off >>= 1)
      sm[r] += __shfl_xor(sm[r], off, 32);
  }
  float inv[8];
#pragma unroll
  for (int r = 0; r < 8; ++r) inv[r] = 1.0f / sm[r];

  // P -> LDS (bf16, A-layout source)
  bf16_t* strip = &lds[wave][0][0];
#pragma unroll
  for (int nt = 0; nt < 32; ++nt) {
#pragma unroll
    for (int r = 0; r < 8; ++r)
      strip[(mb + r) * T_DIM + nt * 16 + n] = (bf16_t)(acc[nt][r] * inv[r]);
  }
  __syncthreads();

  // out = P @ val  (val stored transposed: valT[h][s])
  const bf16_t* vT = valT + (size_t)slice * H_DIM * T_DIM;
  float* op = out + (size_t)slice * T_DIM * H_DIM + (size_t)row0 * H_DIM;
#pragma unroll
  for (int ht = 0; ht < 4; ++ht) {
    f32x8 o = zero8();
#pragma unroll 4
    for (int ks = 0; ks < 16; ++ks) {
      bf16x16 a = load_a_bf16(strip + ks * 32, T_DIM, lane);
      bf16x16 b = load_b_bf16(vT + ht * 16 * T_DIM + ks * 32, T_DIM, lane);
      o = wmma_bf16(a, b, o);
    }
#pragma unroll
    for (int r = 0; r < 8; ++r)
      op[(mb + r) * H_DIM + ht * 16 + n] = o[r];
  }
}

// ---------------------------------------------------------------------------
extern "C" void kernel_launch(void* const* d_in, const int* in_sizes, int n_in,
                              void* d_out, int out_size, void* d_ws, size_t ws_size,
                              hipStream_t stream) {
  (void)in_sizes; (void)n_in; (void)out_size; (void)ws_size;

  const float* idx_i  = (const float*)d_in[0];
  const float* idx_t  = (const float*)d_in[1];
  const float* idx_ir = (const float*)d_in[2];
  const float* Wp  = (const float*)d_in[3];
  const float* bp  = (const float*)d_in[4];
  const float* Wk  = (const float*)d_in[5];
  const float* bk  = (const float*)d_in[6];
  const float* Wpi = (const float*)d_in[7];
  const float* bpi = (const float*)d_in[8];
  const float* Wki = (const float*)d_in[9];
  const float* bki = (const float*)d_in[10];
  const float* Wv  = (const float*)d_in[11];
  const float* bv  = (const float*)d_in[12];
  float* out = (float*)d_out;

  // Workspace layout (bf16): 5 transposed weights (320 KB) + 5 x 8 MiB mats.
  char* w = (char*)d_ws;
  const size_t szWT  = (size_t)5 * H_DIM * A_DIM * sizeof(bf16_t);
  const size_t szMat = (size_t)NSLICE * T_DIM * H_DIM * sizeof(bf16_t);
  bf16_t* WT    = (bf16_t*)(w);
  bf16_t* posb  = (bf16_t*)(w + szWT);
  bf16_t* keyb  = (bf16_t*)(w + szWT + 1 * szMat);
  bf16_t* keyib = (bf16_t*)(w + szWT + 2 * szMat);
  bf16_t* valT  = (bf16_t*)(w + szWT + 3 * szMat);
  bf16_t* wei2b = (bf16_t*)(w + szWT + 4 * szMat);

  wtrans_kernel<<<5, 256, 0, stream>>>(Wp, Wk, Wki, Wv, Wpi, WT);

  proj_kernel<<<dim3(NSLICE, T_DIM / 32), 128, 0, stream>>>(
      idx_i, idx_t, idx_ir, WT, bp, bk, bki, bv, posb, keyb, keyib, valT);

  dim3 grid64(NSLICE, T_DIM / 64);
  pk_pi_kernel<<<grid64, 128, 0, stream>>>(
      posb, keyb, WT + 4 * H_DIM * A_DIM, bpi, wei2b);

  attn_kernel<<<grid64, 128, 0, stream>>>(wei2b, keyib, valT, out);
}
